// Attention_23776938951493
// MI455X (gfx1250) — matrix operand order repaired and checked by reading, not compiled, not run on hardware
//
#include <hip/hip_runtime.h>

typedef __attribute__((ext_vector_type(4)))  _Float16 v4h;
typedef __attribute__((ext_vector_type(8)))  _Float16 v8h;
typedef __attribute__((ext_vector_type(16))) _Float16 v16h;
typedef __attribute__((ext_vector_type(8)))  float    v8f;
typedef __attribute__((ext_vector_type(4)))  unsigned int u32x4;
typedef __attribute__((ext_vector_type(4)))  int i32x4;
typedef __attribute__((ext_vector_type(8)))  int i32x8;

// Detect the TDM builtin (device pass only); fall back to cooperative copies.
#if defined(__AMDGCN__) && defined(__has_builtin)
#if __has_builtin(__builtin_amdgcn_tensor_load_to_lds)
#define HAVE_TDM 1
#endif
#endif
#ifndef HAVE_TDM
#define HAVE_TDM 0
#endif

// D = A(16x32 f16) * B(32x16 f16) + C(16x16 f32)
__device__ __forceinline__ v8f wmma16(v16h a, v16h b, v8f c) {
  return __builtin_amdgcn_wmma_f32_16x16x32_f16(false, a, false, b,
                                                (short)0, c, false, false);
}

// Build a 16-bit A fragment for one 32-wide K window.
// Lane (m = lane%16, half = lane/16) holds K=[half*16,+16) contiguously
// in v0-7. 'row' points at the K-window start.
__device__ __forceinline__ v16h load_a32(const _Float16* row, int half) {
  v8h lo = *(const v8h*)(row + half * 16);
  v8h hi = *(const v8h*)(row + half * 16 + 8);
  v16h r;
#pragma unroll
  for (int i = 0; i < 8; ++i) { r[i] = lo[i]; r[i + 8] = hi[i]; }
  return r;
}

#if HAVE_TDM
// Issue a TDM 2D tile load: global (dim0 x dim1 tensor, row stride stride0,
// elements = 2 bytes) -> LDS, tile t0 x t1, packed rows in LDS.
// D# packing per CDNA5 ISA.
__device__ __forceinline__ void tdm_load_2d(unsigned long long gaddr,
                                            unsigned int lds_addr,
                                            unsigned int dim0, unsigned int dim1,
                                            unsigned long long stride0,
                                            unsigned int t0, unsigned int t1) {
  u32x4 g0;
  g0[0] = 1u;                                            // count=1, user D#
  g0[1] = lds_addr;                                      // LDS byte address
  g0[2] = (unsigned int)gaddr;                           // global_addr[31:0]
  g0[3] = ((unsigned int)(gaddr >> 32) & 0x01ffffffu) |  // global_addr[56:32]
          (2u << 30);                                    // type=2 ("image")
  i32x8 g1;
  g1[0] = (int)(1u << 16);                               // data_size=1 (2B)
  g1[1] = (int)((dim0 & 0xffffu) << 16);                 // tensor_dim0[15:0]
  g1[2] = (int)(((dim0 >> 16) & 0xffffu) | ((dim1 & 0xffffu) << 16));
  g1[3] = (int)(((dim1 >> 16) & 0xffffu) | ((t0 & 0xffffu) << 16)); // tile_dim0
  g1[4] = (int)(t1 & 0xffffu);                           // tile_dim1
  g1[5] = (int)(unsigned int)(stride0 & 0xffffffffu);    // dim0_stride[31:0]
  g1[6] = (int)(unsigned int)((stride0 >> 32) & 0xffffu);
  g1[7] = 0;
  i32x4 z4a = {0, 0, 0, 0}, z4b = {0, 0, 0, 0};          // groups 2/3 unused (2D)
#if __clang_major__ >= 23
  i32x8 z8 = {0, 0, 0, 0, 0, 0, 0, 0};
  __builtin_amdgcn_tensor_load_to_lds(g0, g1, z4a, z4b, z8, 0);
#else
  __builtin_amdgcn_tensor_load_to_lds(g0, g1, z4a, z4b, 0);
#endif
}
#endif

// ---------------------------------------------------------------------------
// Kernel 1: LayerNorm(x) -> f16, one block per token (1024 features)
// ---------------------------------------------------------------------------
__global__ __launch_bounds__(256) void ln_kernel(const float* __restrict__ x,
                                                 const float* __restrict__ gamma,
                                                 const float* __restrict__ beta,
                                                 _Float16* __restrict__ xh) {
  __shared__ float red[8];
  const int t   = blockIdx.x;
  const int tid = threadIdx.x;
  const float* xp = x + (size_t)t * 1024;
  float4 v = *(const float4*)(xp + tid * 4);

  float s = v.x + v.y + v.z + v.w;
#pragma unroll
  for (int off = 16; off >= 1; off >>= 1) s += __shfl_xor(s, off, 32);
  if ((tid & 31) == 0) red[tid >> 5] = s;
  __syncthreads();
  float tot = 0.f;
#pragma unroll
  for (int i = 0; i < 8; ++i) tot += red[i];
  const float mu = tot * (1.0f / 1024.0f);
  __syncthreads();

  float dx = v.x - mu, dy = v.y - mu, dz = v.z - mu, dw = v.w - mu;
  float ss = dx * dx + dy * dy + dz * dz + dw * dw;
#pragma unroll
  for (int off = 16; off >= 1; off >>= 1) ss += __shfl_xor(ss, off, 32);
  if ((tid & 31) == 0) red[tid >> 5] = ss;
  __syncthreads();
  float vtot = 0.f;
#pragma unroll
  for (int i = 0; i < 8; ++i) vtot += red[i];
  const float rstd = rsqrtf(vtot * (1.0f / 1024.0f) + 1e-5f);

  const int c = tid * 4;
  v4h h;
  h[0] = (_Float16)(dx * rstd * gamma[c + 0] + beta[c + 0]);
  h[1] = (_Float16)(dy * rstd * gamma[c + 1] + beta[c + 1]);
  h[2] = (_Float16)(dz * rstd * gamma[c + 2] + beta[c + 2]);
  h[3] = (_Float16)(dw * rstd * gamma[c + 3] + beta[c + 3]);
  *(v4h*)(xh + (size_t)t * 1024 + c) = h;
}

// ---------------------------------------------------------------------------
// Kernel 2: fp32 -> f16 cast (weights)
// ---------------------------------------------------------------------------
__global__ __launch_bounds__(256) void cast_kernel(const float* __restrict__ src,
                                                   _Float16* __restrict__ dst,
                                                   int n4) {
  int i = blockIdx.x * blockDim.x + threadIdx.x;
  if (i < n4) {
    float4 v = ((const float4*)src)[i];
    v4h h;
    h[0] = (_Float16)v.x; h[1] = (_Float16)v.y;
    h[2] = (_Float16)v.z; h[3] = (_Float16)v.w;
    ((v4h*)dst)[i] = h;
  }
}

// ---------------------------------------------------------------------------
// Kernel 3: Y[4096x1024] = X[4096x1024] @ W[1024x1024]^T   (y = x @ W.T)
// 8 waves/block, wave tile 32x32, block tile 64x128.
// ---------------------------------------------------------------------------
__global__ __launch_bounds__(256) void gemm_kernel(const _Float16* __restrict__ X,
                                                   const _Float16* __restrict__ W,
                                                   _Float16* __restrict__ outh,
                                                   float* __restrict__ outf,
                                                   int f32out) {
  const int wave = threadIdx.x >> 5;
  const int lane = threadIdx.x & 31;
  const int mcol = lane & 15;   // A: row within 16-tile; B: column within 16-tile
  const int hlf  = lane >> 4;

  const int row0 = blockIdx.x * 64 + (wave >> 2) * 32;
  const int col0 = blockIdx.y * 128 + (wave & 3) * 32;

  v8f acc[2][2] = {};

#pragma unroll 4
  for (int kk = 0; kk < 32; ++kk) {
    const int kb = kk * 32;
    v16h aA[2], bB[2];
#pragma unroll
    for (int mt = 0; mt < 2; ++mt) {
      const _Float16* arow = X + (size_t)(row0 + mt * 16 + mcol) * 1024 + kb;
      aA[mt] = load_a32(arow, hlf);
      __builtin_prefetch((const void*)(arow + 32), 0, 3);
    }
#pragma unroll
    for (int nt = 0; nt < 2; ++nt) {
      const _Float16* brow =
          W + (size_t)(col0 + nt * 16 + mcol) * 1024 + kb + hlf * 16;
      bB[nt] = *(const v16h*)brow;
      __builtin_prefetch((const void*)(brow + 32), 0, 3);
    }
#pragma unroll
    for (int mt = 0; mt < 2; ++mt)
#pragma unroll
      for (int nt = 0; nt < 2; ++nt)
        acc[mt][nt] = wmma16(aA[mt], bB[nt], acc[mt][nt]);
  }

#pragma unroll
  for (int mt = 0; mt < 2; ++mt)
#pragma unroll
    for (int nt = 0; nt < 2; ++nt)
#pragma unroll
      for (int r = 0; r < 8; ++r) {
        const int row = row0 + mt * 16 + r + 8 * hlf;
        const int col = col0 + nt * 16 + mcol;
        if (f32out) outf[(size_t)row * 1024 + col] = acc[mt][nt][r];
        else        outh[(size_t)row * 1024 + col] = (_Float16)acc[mt][nt][r];
      }
}

// ---------------------------------------------------------------------------
// Kernel 4: per-head transpose of V: Vt[g][d][n] = V[g][n][d]  (g = b*16+h)
// ---------------------------------------------------------------------------
__global__ __launch_bounds__(256) void vtrans_kernel(const _Float16* __restrict__ V,
                                                     _Float16* __restrict__ Vt) {
  int i = blockIdx.x * blockDim.x + threadIdx.x;   // 0 .. 32*131072-1
  int g = i >> 17, rem = i & 131071;
  int d = rem >> 11, n = rem & 2047;
  Vt[i] = V[((size_t)g << 17) + (size_t)n * 64 + d];
}

// ---------------------------------------------------------------------------
// Kernel 5: flash-style attention, one (b*16+h) head per blockIdx.x,
// 64 query rows per block (4 waves x 16 rows), online softmax over 2048 keys.
// K/V tiles staged block-wide into LDS via the Tensor Data Mover.
// ---------------------------------------------------------------------------
__global__ __launch_bounds__(128) void attn_kernel(const _Float16* __restrict__ Q,
                                                   const _Float16* __restrict__ K,
                                                   const _Float16* __restrict__ Vt,
                                                   _Float16* __restrict__ AO) {
  __shared__ __align__(128) _Float16 ldsK[64 * 64];    // 64 keys x 64 dims
  __shared__ __align__(128) _Float16 ldsV[64 * 64];    // 64 dims x 64 keys
  __shared__ __align__(128) _Float16 Pst[4][16 * 64];  // per-wave P staging

  const int g    = blockIdx.x;
  const int wave = threadIdx.x >> 5;
  const int lane = threadIdx.x & 31;
  const int mcol = lane & 15;
  const int hlf  = lane >> 4;
  const int qrow0 = blockIdx.y * 64 + wave * 16;
  const size_t base = (size_t)g << 17;               // * 2048*64
  const float scale = 0.125f;                        // 1/sqrt(64)

  // Q fragments for this wave's 16 rows (K-dim 64 -> two 16x32 A frags)
  const _Float16* qrow = Q + base + (size_t)(qrow0 + mcol) * 64;
  const v16h aQ0 = load_a32(qrow, hlf);
  const v16h aQ1 = load_a32(qrow + 32, hlf);

  v8f cO[4] = {};
  float rmax[8], rsum[8];
#pragma unroll
  for (int r = 0; r < 8; ++r) { rmax[r] = -1e30f; rsum[r] = 0.f; }

#pragma unroll 1
  for (int kc = 0; kc < 32; ++kc) {
    const int key0 = kc * 64;

    __syncthreads();   // previous tile fully consumed before overwrite
#if HAVE_TDM
    if (wave == 0) {
      // K tile: 64 rows (keys) x 64 elems, row stride 64 elems
      tdm_load_2d((unsigned long long)(const void*)(K + base + (size_t)key0 * 64),
                  (unsigned int)(size_t)(void*)ldsK,
                  64u, 2048u, 64ull, 64u, 64u);
      // V tile from Vt: 64 rows (head-dims) x 64 keys, row stride 2048 elems
      tdm_load_2d((unsigned long long)(const void*)(Vt + base + (size_t)key0),
                  (unsigned int)(size_t)(void*)ldsV,
                  2048u, 64u, 2048ull, 64u, 64u);
      __builtin_amdgcn_s_wait_tensorcnt(0);
    }
#else
    {
      const v8h* gk = (const v8h*)(K + base + (size_t)key0 * 64);
#pragma unroll
      for (int i = 0; i < 4; ++i)
        ((v8h*)ldsK)[threadIdx.x + i * 128] = gk[threadIdx.x + i * 128];
#pragma unroll
      for (int i = 0; i < 4; ++i) {
        int c = threadIdx.x + i * 128;       // 512 chunks of 8 halves
        int row = c >> 3, off = (c & 7) * 8;
        ((v8h*)ldsV)[c] =
            *(const v8h*)(Vt + base + (size_t)row * 2048 + key0 + off);
      }
    }
#endif
    __syncthreads();   // tile visible to all waves

    // S(16x64) = Q * K^T : B-lane holds K row (contiguous head-dims) from LDS
    v8f S[4];
#pragma unroll
    for (int nt = 0; nt < 4; ++nt) {
      const _Float16* krow = ldsK + (nt * 16 + mcol) * 64 + hlf * 16;
      v16h bK0 = *(const v16h*)krow;
      v16h bK1 = *(const v16h*)(krow + 32);
      v8f c = {};
      c = wmma16(aQ0, bK0, c);
      c = wmma16(aQ1, bK1, c);
      S[nt] = c;
    }

    // online softmax: row max over this chunk (rows per-VGPR, cols per-lane)
    float tmax[8];
#pragma unroll
    for (int r = 0; r < 8; ++r) {
      float m = S[0][r] * scale;
#pragma unroll
      for (int nt = 1; nt < 4; ++nt) m = fmaxf(m, S[nt][r] * scale);
#pragma unroll
      for (int off = 1; off <= 8; off <<= 1)
        m = fmaxf(m, __shfl_xor(m, off, 16));
      tmax[r] = m;
    }

    float psum[8];
#pragma unroll
    for (int r = 0; r < 8; ++r) {
      const float mnew = fmaxf(rmax[r], tmax[r]);
      const float corr = __expf(rmax[r] - mnew);
      rmax[r] = mnew;
      rsum[r] *= corr;
      psum[r] = 0.f;
#pragma unroll
      for (int dt = 0; dt < 4; ++dt) cO[dt][r] *= corr;
    }

    // P = exp(S - rowmax), stage to LDS in row-major for the A-layout reload
#pragma unroll
    for (int nt = 0; nt < 4; ++nt)
#pragma unroll
      for (int r = 0; r < 8; ++r) {
        float p = __expf(S[nt][r] * scale - rmax[r]);
        psum[r] += p;
        Pst[wave][(r + 8 * hlf) * 64 + nt * 16 + mcol] = (_Float16)p;
      }
#pragma unroll
    for (int r = 0; r < 8; ++r) {
      float s = psum[r];
#pragma unroll
      for (int off = 1; off <= 8; off <<= 1) s += __shfl_xor(s, off, 16);
      rsum[r] += s;
    }

    // wave-internal LDS RAW: CDNA5 split DS counter wait
    asm volatile("s_wait_dscnt 0" ::: "memory");

    const _Float16* prow = &Pst[wave][mcol * 64];
    v16h aP0 = load_a32(prow, hlf);        // keys 0..31
    v16h aP1 = load_a32(prow + 32, hlf);   // keys 32..63

    // O += P * V : B-lane holds an ldsV row (contiguous keys, one head-dim col)
#pragma unroll
    for (int dt = 0; dt < 4; ++dt) {
      const _Float16* vtrow = ldsV + (dt * 16 + mcol) * 64 + hlf * 16;
      v16h bV0 = *(const v16h*)vtrow;
      v16h bV1 = *(const v16h*)(vtrow + 32);
      cO[dt] = wmma16(aP0, bV0, cO[dt]);
      cO[dt] = wmma16(aP1, bV1, cO[dt]);
    }
  }

  // normalize and store (flat [g][row][64] layout == reference flat reshape)
#pragma unroll
  for (int dt = 0; dt < 4; ++dt)
#pragma unroll
    for (int r = 0; r < 8; ++r) {
      float o = cO[dt][r] / rsum[r];
      AO[base + (size_t)(qrow0 + r + 8 * hlf) * 64 + dt * 16 + mcol] =
          (_Float16)o;
    }
}

// ---------------------------------------------------------------------------
extern "C" void kernel_launch(void* const* d_in, const int* in_sizes, int n_in,
                              void* d_out, int out_size, void* d_ws, size_t ws_size,
                              hipStream_t stream) {
  (void)in_sizes; (void)n_in; (void)out_size; (void)ws_size;
  const float* x     = (const float*)d_in[0];
  const float* gamma = (const float*)d_in[1];
  const float* beta  = (const float*)d_in[2];
  const float* Wq    = (const float*)d_in[3];
  const float* Wk    = (const float*)d_in[4];
  const float* Wv    = (const float*)d_in[5];
  const float* Wo    = (const float*)d_in[6];

  constexpr size_t NTOK = 4096;   // B*N
  constexpr size_t DM   = 1024;

  _Float16* xh  = (_Float16*)d_ws;            // 4096x1024
  _Float16* Wqh = xh  + NTOK * DM;            // 1024x1024 each
  _Float16* Wkh = Wqh + DM * DM;
  _Float16* Wvh = Wkh + DM * DM;
  _Float16* Woh = Wvh + DM * DM;
  _Float16* Qh  = Woh + DM * DM;              // 4096x1024 each
  _Float16* Kh  = Qh  + NTOK * DM;
  _Float16* Vh  = Kh  + NTOK * DM;
  _Float16* Vt  = Vh  + NTOK * DM;
  _Float16* AO  = Vt  + NTOK * DM;

  ln_kernel<<<4096, 256, 0, stream>>>(x, gamma, beta, xh);

  const int n4 = (int)(DM * DM / 4);
  cast_kernel<<<1024, 256, 0, stream>>>(Wq, Wqh, n4);
  cast_kernel<<<1024, 256, 0, stream>>>(Wk, Wkh, n4);
  cast_kernel<<<1024, 256, 0, stream>>>(Wv, Wvh, n4);
  cast_kernel<<<1024, 256, 0, stream>>>(Wo, Woh, n4);

  dim3 ggrid(64, 8);
  gemm_kernel<<<ggrid, 256, 0, stream>>>(xh, Wqh, Qh, nullptr, 0);
  gemm_kernel<<<ggrid, 256, 0, stream>>>(xh, Wkh, Kh, nullptr, 0);
  gemm_kernel<<<ggrid, 256, 0, stream>>>(xh, Wvh, Vh, nullptr, 0);

  vtrans_kernel<<<16384, 256, 0, stream>>>(Vh, Vt);

  attn_kernel<<<dim3(32, 32), 128, 0, stream>>>(Qh, Kh, Vt, AO);

  gemm_kernel<<<ggrid, 256, 0, stream>>>(AO, Woh, nullptr, (float*)d_out, 1);
}
